// LlamaAttention_51539608294
// MI455X (gfx1250) — matrix operand order repaired and checked by reading, not compiled, not run on hardware
//
#include <hip/hip_runtime.h>

// ---------------------------------------------------------------------------
// LlamaAttention on MI455X (gfx1250, wave32, WMMA bf16 16x16x32 f32-accum)
//   B=2, S=2048, D=2048, H=32, KVH=8, hd=64, KV_DIM=512, rep=4
// Pipeline: cvt x->bf16, cvt+transpose weights, WMMA GEMMs for Q/K/V
// (B-tile staged to LDS by the Tensor Data Mover, double buffered,
//  TENSORcnt-pipelined), fused flash attention (online softmax),
// WMMA GEMM for wo -> f32.
// ---------------------------------------------------------------------------

typedef __attribute__((ext_vector_type(16))) __bf16   v16bf;
typedef __attribute__((ext_vector_type(8)))  __bf16   v8bf;
typedef __attribute__((ext_vector_type(8)))  float    v8f;
typedef __attribute__((ext_vector_type(4)))  uint32_t u32x4;
typedef __attribute__((ext_vector_type(8)))  uint32_t u32x8;

#define WMMA_BF16(a, b, c) \
  __builtin_amdgcn_wmma_f32_16x16x32_bf16(false, (a), false, (b), (short)0, (c), false, false)

static constexpr int BATCH   = 2;
static constexpr int SEQ     = 2048;
static constexpr int DIM     = 2048;
static constexpr int HEADS   = 32;
static constexpr int KVHEADS = 8;
static constexpr int HD      = 64;
static constexpr int KV_DIM  = 512;
static constexpr int TOKENS  = BATCH * SEQ;

// ---- A-operand fragment: 16x32 bf16, k = 16*half + s -------------------
// lane m (0..15 per half): VGPR0..7 = K[half*16 .. +15] (contiguous, matches B)
__device__ __forceinline__ v16bf load_a16x32(const __bf16* base, int stride) {
  const int lane = threadIdx.x & 31;
  const int m = lane & 15, half = lane >> 4;
  const __bf16* p = base + (size_t)m * stride + half * 16;
  v8bf lo = *(const v8bf*)(p);
  v8bf hi = *(const v8bf*)(p + 8);
  return __builtin_shufflevector(lo, hi, 0,1,2,3,4,5,6,7,8,9,10,11,12,13,14,15);
}

// ---- B-operand fragment: 32x16 bf16, k = 16*half + s -------------------
// lane n = column, lane-half selects K range; 16 contiguous K values per lane.
__device__ __forceinline__ v16bf load_b32x16(const __bf16* baseT, int stride) {
  const int lane = threadIdx.x & 31;
  const int n = lane & 15, half = lane >> 4;
  const __bf16* p = baseT + (size_t)n * stride + half * 16;
  return *(const v16bf*)(p);          // 32B contiguous per lane
}

// B fragment out of an LDS tile laid out row-major [64 cols][32 k]
__device__ __forceinline__ v16bf load_b_lds(const __bf16* buf, int jcol) {
  const int lane = threadIdx.x & 31;
  const int n = lane & 15, half = lane >> 4;
  const __bf16* p = buf + (jcol * 16 + n) * 32 + half * 16;
  return *(const v16bf*)(p);          // 2x ds_load_b128
}

// ---- Tensor Data Mover: DMA one 64x32 bf16 tile (4KB) global -> LDS ------
// 2D descriptor (CDNA5 ISA D# layout): dim0 = 32 contiguous bf16 along K,
// dim1 = 64 rows with stride Kstride elements. One instruction per tile,
// tracked with TENSORcnt.
__device__ __forceinline__ void tdm_stage_btile(const __bf16* gtile,
                                                uint32_t lds_off, int Kstride) {
  const uint64_t ga  = (uint64_t)(uintptr_t)gtile;
  const uint32_t td0 = (uint32_t)Kstride;   // dim0 extent (elements), never OOB
  const uint32_t td1 = 0x00100000u;         // dim1 extent: large, never OOB

  u32x4 g0;
  g0[0] = 1u;                                             // count=1 (valid)
  g0[1] = lds_off;                                        // lds_addr (bytes)
  g0[2] = (uint32_t)ga;                                   // global_addr[31:0]
  g0[3] = (uint32_t)((ga >> 32) & 0x01FFFFFFu) | (2u << 30);  // addr[56:32]|type=2

  u32x8 g1;
  g1[0] = 1u << 16;                                       // data_size=1 (2B)
  g1[1] = (td0 & 0xFFFFu) << 16;                          // tensor_dim0[15:0]
  g1[2] = (td0 >> 16) | ((td1 & 0xFFFFu) << 16);          // td0[31:16] | td1[15:0]
  g1[3] = (td1 >> 16) | (32u << 16);                      // td1[31:16] | tile_dim0=32
  g1[4] = 64u;                                            // tile_dim1=64, tile_dim2=0
  g1[5] = (uint32_t)Kstride;                              // tensor_dim0_stride[31:0]
  g1[6] = 0u;                                             // stride0[47:32] | stride1[15:0]
  g1[7] = 0u;                                             // tensor_dim1_stride[47:16]

  u32x4 gz = {0u, 0u, 0u, 0u};                            // groups 2/3: unused (2D)
  asm volatile("tensor_load_to_lds %0, %1, %2, %3"
               :: "s"(g0), "s"(g1), "s"(gz), "s"(gz)
               : "memory");
}

// ---------------------------------------------------------------------------
__global__ void cvt_bf16_kernel(const float* __restrict__ src,
                                __bf16* __restrict__ dst, int n) {
  for (int i = blockIdx.x * blockDim.x + threadIdx.x; i < n; i += gridDim.x * blockDim.x)
    dst[i] = (__bf16)src[i];
}

__global__ void cvt_transpose_kernel(const float* __restrict__ w,
                                     __bf16* __restrict__ wt, int K, int N) {
  int total = K * N;
  for (int i = blockIdx.x * blockDim.x + threadIdx.x; i < total; i += gridDim.x * blockDim.x) {
    int k = i / N, n = i - k * N;
    wt[(size_t)n * K + k] = (__bf16)w[i];
  }
}

// ---------------------------------------------------------------------------
// WMMA GEMM: C[M,N] = A[M,K] @ B[K,N]; A bf16 row-major, BT bf16 [N][K].
// Block = 128 threads (4 waves): block tile 128x64, wave tile 32x64
// (2 A-frags x 4 B-frags = 8 WMMA per 32-deep k-step).
// B tile staged to LDS by the TDM (one tensor_load_to_lds per 4KB tile),
// double buffered and shared by all 4 waves; TENSORcnt <=1 keeps the next
// tile in flight while the current one feeds the WMMAs.
// mode 0: bf16 row-major; mode 1: f32 row-major; mode 2: bf16 V-transposed.
__global__ __launch_bounds__(128) void gemm_wmma_kernel(
    const __bf16* __restrict__ A, const __bf16* __restrict__ BT,
    void* __restrict__ Cout, int M, int N, int K, int mode) {
  __shared__ __align__(16) __bf16 btile[2][64 * 32];   // 2 x 4KB ring

  const int wave = threadIdx.x >> 5;
  const int row0 = blockIdx.y * 128 + wave * 32;
  const int col0 = blockIdx.x * 64;
  const bool issuer = (wave == 0);   // TDM ignores EXEC; issue from one wave

  v8f acc[2][4] = {};
  const __bf16* arow0 = A + (size_t)row0 * K;
  const __bf16* arow1 = arow0 + (size_t)16 * K;
  const __bf16* btcol = BT + (size_t)col0 * K;

  if (issuer)
    tdm_stage_btile(btcol, (uint32_t)(size_t)&btile[0][0], K);   // tile 0 in flight

  int parity = 0;
  for (int k0 = 0; k0 < K; k0 += 32) {
    if (k0 + 32 < K) {
      if (issuer)
        tdm_stage_btile(btcol + k0 + 32, (uint32_t)(size_t)&btile[parity ^ 1][0], K);
      __builtin_amdgcn_s_wait_tensorcnt(1);            // current tile done
    } else {
      __builtin_amdgcn_s_wait_tensorcnt(0);
    }
    __syncthreads();                                   // tile visible to all waves

    __builtin_prefetch(arow0 + k0 + 64, 0, 1);
    v16bf a0 = load_a16x32(arow0 + k0, K);
    v16bf a1 = load_a16x32(arow1 + k0, K);
    v16bf b0 = load_b_lds(btile[parity], 0);
    v16bf b1 = load_b_lds(btile[parity], 1);
    v16bf b2 = load_b_lds(btile[parity], 2);
    v16bf b3 = load_b_lds(btile[parity], 3);
    acc[0][0] = WMMA_BF16(a0, b0, acc[0][0]);
    acc[0][1] = WMMA_BF16(a0, b1, acc[0][1]);
    acc[0][2] = WMMA_BF16(a0, b2, acc[0][2]);
    acc[0][3] = WMMA_BF16(a0, b3, acc[0][3]);
    acc[1][0] = WMMA_BF16(a1, b0, acc[1][0]);
    acc[1][1] = WMMA_BF16(a1, b1, acc[1][1]);
    acc[1][2] = WMMA_BF16(a1, b2, acc[1][2]);
    acc[1][3] = WMMA_BF16(a1, b3, acc[1][3]);

    __syncthreads();                                   // reads done before ring reuse
    parity ^= 1;
  }

  const int lane = threadIdx.x & 31;
  const int half = lane >> 4, cl = lane & 15;
#pragma unroll
  for (int sub = 0; sub < 2; ++sub)
#pragma unroll
    for (int j = 0; j < 4; ++j)
#pragma unroll
      for (int i = 0; i < 8; ++i) {
        int r = row0 + sub * 16 + i + 8 * half;        // C layout row mapping
        int c = col0 + j * 16 + cl;
        float v = acc[sub][j][i];
        if (mode == 0) {
          ((__bf16*)Cout)[(size_t)r * N + c] = (__bf16)v;
        } else if (mode == 1) {
          ((float*)Cout)[(size_t)r * N + c] = v;
        } else {                                       // V transposed for PV B-operand
          int b = r >> 11;                             // SEQ = 2048
          int s = r & (SEQ - 1);
          int kvh = c >> 6, hd = c & 63;
          ((__bf16*)Cout)[((size_t)((b * KVHEADS + kvh) * HD + hd)) * SEQ + s] = (__bf16)v;
        }
      }
}

// ---------------------------------------------------------------------------
// Fused flash attention: one wave per (16-query tile, head, batch).
__global__ __launch_bounds__(32) void flash_attn_kernel(
    const __bf16* __restrict__ Q, const __bf16* __restrict__ Kc,
    const __bf16* __restrict__ Vt, __bf16* __restrict__ Oa) {
  __shared__ __align__(16) __bf16 pbuf[16 * 32];       // P: C-layout -> A-layout staging

  const int q0  = blockIdx.x * 16;
  const int h   = blockIdx.y;
  const int b   = blockIdx.z;
  const int kvh = h >> 2;                              // rep = 4

  const int lane = threadIdx.x & 31;
  const int half = lane >> 4, cl = lane & 15;

  const __bf16* qbase = Q  + ((size_t)(b * SEQ + q0)) * DIM + h * HD;
  const __bf16* kbase = Kc + ((size_t)(b * SEQ)) * KV_DIM + kvh * HD;
  const __bf16* vbase = Vt + ((size_t)((b * KVHEADS + kvh) * HD)) * SEQ;

  v16bf qf0 = load_a16x32(qbase,      DIM);            // Q tile 16x64, loaded once
  v16bf qf1 = load_a16x32(qbase + 32, DIM);

  float mrow[8], lrow[8];
  v8f   o[4] = {};
#pragma unroll
  for (int i = 0; i < 8; ++i) { mrow[i] = -1e30f; lrow[i] = 0.0f; }

  for (int kb0 = 0; kb0 < SEQ; kb0 += 32) {
    // ---- S = (Q K^T) * scale : batch all 4 K-frags, then 4 WMMAs ----
    const __bf16* kr0 = kbase + (size_t)kb0 * KV_DIM;
    const __bf16* kr1 = kbase + (size_t)(kb0 + 16) * KV_DIM;
    __builtin_prefetch(kr0 + 32 * KV_DIM, 0, 1);
    v16bf kf0 = load_b32x16(kr0,      KV_DIM);
    v16bf kf1 = load_b32x16(kr0 + 32, KV_DIM);
    v16bf kf2 = load_b32x16(kr1,      KV_DIM);
    v16bf kf3 = load_b32x16(kr1 + 32, KV_DIM);
    v8f s[2] = {};
    s[0] = WMMA_BF16(qf0, kf0, s[0]);
    s[0] = WMMA_BF16(qf1, kf1, s[0]);
    s[1] = WMMA_BF16(qf0, kf2, s[1]);
    s[1] = WMMA_BF16(qf1, kf3, s[1]);
#pragma unroll
    for (int nt = 0; nt < 2; ++nt)
#pragma unroll
      for (int i = 0; i < 8; ++i) s[nt][i] *= 0.125f;  // 1/sqrt(64)

    // ---- online softmax row stats (row i+8*half lives on this lane-half) ----
    float fac[8];
#pragma unroll
    for (int i = 0; i < 8; ++i) {
      float cm = fmaxf(s[0][i], s[1][i]);
#pragma unroll
      for (int d = 1; d < 16; d <<= 1) cm = fmaxf(cm, __shfl_xor(cm, d, 32));
      float mnew = fmaxf(mrow[i], cm);
      fac[i] = __expf(mrow[i] - mnew);
      mrow[i] = mnew;
    }
#pragma unroll
    for (int i = 0; i < 8; ++i) lrow[i] *= fac[i];
#pragma unroll
    for (int j = 0; j < 4; ++j)
#pragma unroll
      for (int i = 0; i < 8; ++i) o[j][i] *= fac[i];

    // ---- P = exp(S - m); stage to LDS row-major for A-fragment reload ----
    float psum[8];
#pragma unroll
    for (int i = 0; i < 8; ++i) psum[i] = 0.0f;
#pragma unroll
    for (int nt = 0; nt < 2; ++nt)
#pragma unroll
      for (int i = 0; i < 8; ++i) {
        float p = __expf(s[nt][i] - mrow[i]);
        psum[i] += p;
        pbuf[(i + 8 * half) * 32 + nt * 16 + cl] = (__bf16)p;
      }
#pragma unroll
    for (int i = 0; i < 8; ++i) {
      float t = psum[i];
#pragma unroll
      for (int d = 1; d < 16; d <<= 1) t += __shfl_xor(t, d, 32);
      lrow[i] += t;
    }

    // ---- O += P @ V : batch all 4 V-frags, then 4 WMMAs ----
    v16bf pf  = load_a16x32(pbuf, 32);                 // same-wave DS ops in order
    v16bf vf0 = load_b32x16(vbase + (size_t)( 0) * SEQ + kb0, SEQ);
    v16bf vf1 = load_b32x16(vbase + (size_t)(16) * SEQ + kb0, SEQ);
    v16bf vf2 = load_b32x16(vbase + (size_t)(32) * SEQ + kb0, SEQ);
    v16bf vf3 = load_b32x16(vbase + (size_t)(48) * SEQ + kb0, SEQ);
    o[0] = WMMA_BF16(pf, vf0, o[0]);
    o[1] = WMMA_BF16(pf, vf1, o[1]);
    o[2] = WMMA_BF16(pf, vf2, o[2]);
    o[3] = WMMA_BF16(pf, vf3, o[3]);
  }

  // ---- normalize and store bf16 [tok][D] for the wo GEMM ----
#pragma unroll
  for (int j = 0; j < 4; ++j)
#pragma unroll
    for (int i = 0; i < 8; ++i) {
      int r = q0 + i + 8 * half;
      int c = h * HD + j * 16 + cl;
      Oa[((size_t)(b * SEQ + r)) * DIM + c] = (__bf16)(o[j][i] / lrow[i]);
    }
}

// ---------------------------------------------------------------------------
extern "C" void kernel_launch(void* const* d_in, const int* in_sizes, int n_in,
                              void* d_out, int out_size, void* d_ws, size_t ws_size,
                              hipStream_t stream) {
  const float* x  = (const float*)d_in[0];
  const float* wq = (const float*)d_in[1];
  const float* wk = (const float*)d_in[2];
  const float* wv = (const float*)d_in[3];
  const float* wo = (const float*)d_in[4];

  char* ws = (char*)d_ws;
  size_t off = 0;
  auto alloc = [&](size_t bytes) { char* p = ws + off; off += (bytes + 255) & ~size_t(255); return p; };

  __bf16* xb   = (__bf16*)alloc((size_t)TOKENS * DIM * 2);
  __bf16* wqT  = (__bf16*)alloc((size_t)DIM * DIM * 2);
  __bf16* wkT  = (__bf16*)alloc((size_t)KV_DIM * DIM * 2);
  __bf16* wvT  = (__bf16*)alloc((size_t)KV_DIM * DIM * 2);
  __bf16* woT  = (__bf16*)alloc((size_t)DIM * DIM * 2);
  __bf16* qb   = (__bf16*)alloc((size_t)TOKENS * DIM * 2);
  __bf16* kb   = (__bf16*)alloc((size_t)TOKENS * KV_DIM * 2);
  __bf16* vT   = (__bf16*)alloc((size_t)TOKENS * KV_DIM * 2);
  __bf16* attn = (__bf16*)alloc((size_t)TOKENS * DIM * 2);
  // total ~76 MiB of workspace

  cvt_bf16_kernel<<<2048, 256, 0, stream>>>(x, xb, TOKENS * DIM);
  cvt_transpose_kernel<<<2048, 256, 0, stream>>>(wq, wqT, DIM, DIM);
  cvt_transpose_kernel<<<1024, 256, 0, stream>>>(wk, wkT, DIM, KV_DIM);
  cvt_transpose_kernel<<<1024, 256, 0, stream>>>(wv, wvT, DIM, KV_DIM);
  cvt_transpose_kernel<<<2048, 256, 0, stream>>>(wo, woT, DIM, DIM);

  gemm_wmma_kernel<<<dim3(DIM / 64, TOKENS / 128), 128, 0, stream>>>(
      xb, wqT, qb, TOKENS, DIM, DIM, /*mode=*/0);
  gemm_wmma_kernel<<<dim3(KV_DIM / 64, TOKENS / 128), 128, 0, stream>>>(
      xb, wkT, kb, TOKENS, KV_DIM, DIM, /*mode=*/0);
  gemm_wmma_kernel<<<dim3(KV_DIM / 64, TOKENS / 128), 128, 0, stream>>>(
      xb, wvT, vT, TOKENS, KV_DIM, DIM, /*mode=*/2);

  flash_attn_kernel<<<dim3(SEQ / 16, HEADS, BATCH), 32, 0, stream>>>(qb, kb, vT, attn);

  gemm_wmma_kernel<<<dim3(DIM / 64, TOKENS / 128), 128, 0, stream>>>(
      attn, woT, d_out, TOKENS, DIM, DIM, /*mode=*/1);
}
